// SupConLoss_74199855006112
// MI455X (gfx1250) — matrix operand order repaired and checked by reading, not compiled, not run on hardware
//
#include <hip/hip_runtime.h>
#include <hip/hip_bf16.h>

typedef __attribute__((ext_vector_type(4)))  _Float16 h4;
typedef __attribute__((ext_vector_type(8)))  _Float16 v8h;
typedef __attribute__((ext_vector_type(16))) _Float16 v16h;
typedef __attribute__((ext_vector_type(8)))  float    v8f;

#define NB   4096      // batch
#define DD   8192      // feature dim
#define PC   112       // classes (100) padded to multiple of 16
#define TAUI 0.1f      // 1/tau

// ---------------------------------------------------------------------------
// Kernel 1: stream embeddings (128 MB, HBM-bound), produce per-row:
//   denom[i] = sum_d exp(x^2/tau) - A[i,i]
//   Bdiag[i] = exp(A[i,i]/tau)
//   Bh[i][j] = (f16) exp(A[i,j]/tau)  for j < NB   (32 MB, stays in L2)
// ---------------------------------------------------------------------------
__global__ void k_rows(const float* __restrict__ emb,
                       _Float16* __restrict__ Bh,
                       float* __restrict__ denom,
                       float* __restrict__ Bdiag) {
  const int row = blockIdx.x;
  const int tid = threadIdx.x;
  const float4* rp = (const float4*)(emb + (size_t)row * DD);

  float acc = 0.f;
  float adiag = 0.f;
  bool  hit = false;

#pragma unroll
  for (int it = 0; it < DD / (256 * 4); ++it) {
    const int v = it * 256 + tid;       // float4 index within the row
    const float4 x = rp[v];
    const float a0 = __expf(x.x * x.x * TAUI);
    const float a1 = __expf(x.y * x.y * TAUI);
    const float a2 = __expf(x.z * x.z * TAUI);
    const float a3 = __expf(x.w * x.w * TAUI);
    acc += (a0 + a1) + (a2 + a3);

    const int base = v * 4;
    if (row >= base && row < base + 4) {
      adiag = (row == base) ? a0 : (row == base + 1) ? a1
            : (row == base + 2) ? a2 : a3;
      hit = true;
    }
    if (base < NB) {
      h4 b;
      b[0] = (_Float16)__expf(a0 * TAUI);
      b[1] = (_Float16)__expf(a1 * TAUI);
      b[2] = (_Float16)__expf(a2 * TAUI);
      b[3] = (_Float16)__expf(a3 * TAUI);
      *(h4*)(Bh + (size_t)row * NB + base) = b;   // 8-byte store
    }
  }

  __shared__ float sred[256];
  __shared__ float sdiag;
  sred[tid] = acc;
  if (hit) sdiag = adiag;               // exactly one thread per block hits
  __syncthreads();
  for (int s = 128; s > 0; s >>= 1) {
    if (tid < s) sred[tid] += sred[tid + s];
    __syncthreads();
  }
  if (tid == 0) {
    denom[row] = sred[0] - sdiag;
    Bdiag[row] = __expf(sdiag * TAUI);
  }
}

// ---------------------------------------------------------------------------
// Kernel 2: one-hot class matrix, transposed: Et[c][j] = (labels[j]==c)
// Layout [PC][NB] f16 so each lane's WMMA B-fragment is two contiguous 8-half runs.
// ---------------------------------------------------------------------------
__global__ void k_onehot(const int* __restrict__ labels,
                         _Float16* __restrict__ Et) {
  const int j = blockIdx.x * 256 + threadIdx.x;
  const int c = blockIdx.y;
  const int lab = labels[j];
  Et[(size_t)c * NB + j] = (_Float16)((lab == c) ? 1.0f : 0.0f);
}

__global__ void k_hist(const int* __restrict__ labels, int* __restrict__ hist) {
  const int j = blockIdx.x * 256 + threadIdx.x;
  atomicAdd(&hist[labels[j]], 1);
}

// ---------------------------------------------------------------------------
// Kernel 3: M[i][c] = sum_j Bh[i][j] * Et[c][j]  via v_wmma_f32_16x16x32_f16.
// One wave per 16x16 output tile; K = 4096 (128 WMMAs). Bh (32 MB) is
// L2-resident, so this is L2-bandwidth bound and tiny vs. kernel 1.
//
// A-fragment (16x32 f16, MxK), per ISA layout:
//   lane L: row m = L%16;  v0..v3 hold K = (L/16)*8 + 0..7,
//                          v4..v7 hold K = (L/16)*8 + 16..23
// B-fragment (32x16 f16, KxN):
//   lane L: col c = L%16;  v0..v3 hold K = (L/16)*8 + 0..7,
//                          v4..v7 hold K = (L/16)*8 + 16..23
// ---------------------------------------------------------------------------
__global__ void k_gemm(const _Float16* __restrict__ Bh,
                       const _Float16* __restrict__ Et,
                       float* __restrict__ M) {
  const int wave = threadIdx.x >> 5;            // 4 waves / block
  const int lane = threadIdx.x & 31;
  const int tile = blockIdx.x * 4 + wave;       // 0..1791
  const int iTile = tile & 255;                 // 256 row tiles
  const int cTile = tile >> 8;                  // 7 class tiles
  const int iBase = iTile * 16;
  const int cBase = cTile * 16;
  const int m  = lane & 15;
  const int hi = lane >> 4;                     // 0 or 1

  const _Float16* aRow = Bh + (size_t)(iBase + m) * NB + hi * 8;
  const _Float16* bCol = Et + (size_t)(cBase + m) * NB + hi * 8;

  v8f acc = {};
#pragma unroll 4
  for (int k = 0; k < NB; k += 32) {
    union { v16h v; v8h h[2]; } af, bf;
    af.h[0] = *(const v8h*)(aRow + k);          // K = hi*8 + 0..7
    af.h[1] = *(const v8h*)(aRow + k + 16);     // K = hi*8 + 16..23
    bf.h[0] = *(const v8h*)(bCol + k);          // K = hi*8 + 0..7
    bf.h[1] = *(const v8h*)(bCol + k + 16);     // K = hi*8 + 16..23
    acc = __builtin_amdgcn_wmma_f32_16x16x32_f16(
        false, af.v, false, bf.v, (short)0, acc, false, false);
  }

  // D layout: lane L holds rows (L/16)*8 + r (r=0..7), col L%16
  float* outp = M + (size_t)(iBase + hi * 8) * PC + (cBase + m);
#pragma unroll
  for (int r = 0; r < 8; ++r) outp[(size_t)r * PC] = acc[r];
}

// ---------------------------------------------------------------------------
// Kernel 4: loss = sum_i log( ((M[i,lab]-Bdiag[i])/denom[i]) / (hist[lab]-1) )
// ---------------------------------------------------------------------------
__global__ void k_final(const float* __restrict__ M,
                        const float* __restrict__ denom,
                        const float* __restrict__ Bdiag,
                        const int* __restrict__ labels,
                        const int* __restrict__ hist,
                        float* __restrict__ out) {
  const int i = blockIdx.x * 256 + threadIdx.x;
  const int lab = labels[i];
  const float suma = (M[(size_t)i * PC + lab] - Bdiag[i]) / denom[i];
  const float cnt = (float)(hist[lab] - 1);
  const float val = __logf(suma / cnt);

  __shared__ float sred[256];
  sred[threadIdx.x] = val;
  __syncthreads();
  for (int s = 128; s > 0; s >>= 1) {
    if (threadIdx.x < s) sred[threadIdx.x] += sred[threadIdx.x + s];
    __syncthreads();
  }
  if (threadIdx.x == 0) atomicAdd(out, sred[0]);
}

// ---------------------------------------------------------------------------
extern "C" void kernel_launch(void* const* d_in, const int* in_sizes, int n_in,
                              void* d_out, int out_size, void* d_ws, size_t ws_size,
                              hipStream_t stream) {
  const float* emb    = (const float*)d_in[0];
  const int*   labels = (const int*)d_in[1];
  float*       out    = (float*)d_out;
  char*        ws     = (char*)d_ws;

  // Workspace layout (all 16B-aligned)
  const size_t offB    = 0;                                    // Bh: 32 MiB
  const size_t offDen  = offB   + (size_t)NB * NB * 2;         // denom: 16 KiB
  const size_t offBd   = offDen + (size_t)NB * 4;              // Bdiag: 16 KiB
  const size_t offHist = offBd  + (size_t)NB * 4;              // hist: 512 B
  const size_t offEt   = offHist + 512;                        // Et: 896 KiB
  const size_t offM    = offEt  + (size_t)PC * NB * 2;         // M: 1.75 MiB

  _Float16* Bh    = (_Float16*)(ws + offB);
  float*    denom = (float*)(ws + offDen);
  float*    Bdiag = (float*)(ws + offBd);
  int*      hist  = (int*)(ws + offHist);
  _Float16* Et    = (_Float16*)(ws + offEt);
  float*    M     = (float*)(ws + offM);

  hipMemsetAsync(hist, 0, 512, stream);
  hipMemsetAsync(out, 0, sizeof(float), stream);

  k_rows<<<NB, 256, 0, stream>>>(emb, Bh, denom, Bdiag);
  k_onehot<<<dim3(NB / 256, PC), 256, 0, stream>>>(labels, Et);
  k_hist<<<NB / 256, 256, 0, stream>>>(labels, hist);
  k_gemm<<<(256 * (PC / 16)) / 4, 128, 0, stream>>>(Bh, Et, M);
  k_final<<<NB / 256, 256, 0, stream>>>(M, denom, Bdiag, labels, hist, out);
}
